// OutputLayersSymm_29772713295881
// MI455X (gfx1250) — matrix operand order repaired and checked by reading, not compiled, not run on hardware
//
#include <hip/hip_runtime.h>

typedef __attribute__((ext_vector_type(16))) _Float16 v16h;
typedef __attribute__((ext_vector_type(8)))  float    v8f;
typedef __attribute__((ext_vector_type(4)))  int      vint4;

#define LSEQ    128
#define NBATCH  4
#define DFEAT   768
#define WSTRIP  13                 // interior: 0..3 left edge, 4..8 interior, 9..12 right edge
#define PPI     (LSEQ*WSTRIP)      // 1664 interior pixels per image
#define PXTOT   (NBATCH*PPI)       // 6656 interior pixels per stack
#define PXTILES (PXTOT/16)         // 416
#define HP      (LSEQ+2)           // 130 padded rows
#define WP      (WSTRIP+2)         // 15 padded cols
#define PXPADI  (HP*WP)            // 1950 padded pixels per image
#define PXPADS  (NBATCH*PXPADI)    // 7800 padded pixels per stack

union FragH { vint4 q[2]; v16h h; };

// ---------------------------------------------------------------------------
// Zero-fill workspace span (halo zeros for all strip buffers), 16B per thread.
__global__ void k_fill_zero(vint4* __restrict__ p, int n16)
{
    int idx = blockIdx.x * blockDim.x + threadIdx.x;
    if (idx >= n16) return;
    vint4 z = {0, 0, 0, 0};
    p[idx] = z;
}

// ---------------------------------------------------------------------------
// Build strip input interior: strip[b][i+1][jc+1][c] = x[i][b][c]  (f32 -> f16)
// batch layout is (L, N, D); strip is halo-padded (HP x WP).
__global__ void k_build_strip(const float* __restrict__ x,
                              _Float16* __restrict__ strip, int total)
{
    int idx = blockIdx.x * blockDim.x + threadIdx.x;
    if (idx >= total) return;
    int c  = idx % DFEAT;
    int t  = idx / DFEAT;
    int jc = t % WSTRIP;
    int t2 = t / WSTRIP;
    int i  = t2 % LSEQ;
    int b  = t2 / LSEQ;
    strip[((size_t)(b * HP + i + 1) * WP + jc + 1) * DFEAT + c] =
        (_Float16)x[(i * NBATCH + b) * DFEAT + c];
}

// ---------------------------------------------------------------------------
// Pack conv weights (3,3,cinFull,Cout) f32 into WMMA-B fragment order, f16.
// Layout: [tap][kc][nc][lane*16+e], with k = (e&7) + (lane&16 ? 8 : 0) + (e&8 ? 16 : 0), n = lane&15.
// transpose=1 swaps the spatial taps (column stack). ciOff selects channel slice.
__global__ void k_pack_w(const float* __restrict__ w, _Float16* __restrict__ pb,
                         int cinFull, int ciOff, int KC, int NC, int Cout,
                         int transpose, int total)
{
    int idx = blockIdx.x * blockDim.x + threadIdx.x;
    if (idx >= total) return;
    int e    = idx & 15;
    int lane = (idx >> 4) & 31;
    int t    = idx >> 9;
    int nc   = t % NC;
    int kc   = (t / NC) % KC;
    int tap  = t / (NC * KC);
    int a    = tap / 3, bb = tap % 3;
    int stap = transpose ? (bb * 3 + a) : tap;
    int k    = (e & 7) + ((lane & 16) ? 8 : 0) + ((e & 8) ? 16 : 0);
    int ci   = ciOff + kc * 32 + k;
    int co   = nc * 16 + (lane & 15);
    pb[idx] = (_Float16)w[((size_t)(stap * cinFull + ci)) * Cout + co];
}

// ---------------------------------------------------------------------------
// 3x3 SAME conv on halo-padded strips as tap-summed WMMA GEMM. One wave owns a
// 16-pixel x 64-cout tile (4 accumulators reuse each A fragment). All loads are
// unconditional (zero halo), A is double-buffered so WMMAs only wait on the
// L2-resident B stream.
template<int CIN, int COUT>
__global__ void __launch_bounds__(256)
k_conv3x3_wmma(const _Float16* __restrict__ in, const _Float16* __restrict__ pb,
               const float* __restrict__ bias, _Float16* __restrict__ out, int tiles)
{
    constexpr int KC = CIN / 32;
    constexpr int NC = COUT / 16;
    int wave = blockIdx.x * (blockDim.x >> 5) + (threadIdx.x >> 5);
    if (wave >= tiles) return;                    // wave-uniform
    int lane  = threadIdx.x & 31;
    int co64  = wave / PXTILES;
    int pxt   = wave - co64 * PXTILES;
    int p0    = pxt << 4;                         // interior pixel base of tile
    int b     = p0 / PPI;
    int pin   = p0 - b * PPI;
    int m     = lane & 15;
    int pm    = pin + m;
    int i0    = pm / WSTRIP;
    int jc0   = pm - i0 * WSTRIP;
    int khalf = (lane & 16) ? 8 : 0;

    v8f zero = {};
    v8f acc[4] = {zero, zero, zero, zero};

    for (int tap = 0; tap < 9; ++tap) {
        int di = tap / 3;                          // padded row = i0 + di  (di in 0..2)
        int dj = tap - di * 3;                     // padded col = jc0 + dj
        const _Float16* row  = in + ((size_t)((b * HP + i0 + di) * WP + jc0 + dj)) * CIN + khalf;
        const _Float16* wtap = pb + (((size_t)tap * KC * NC + (co64 << 2)) << 9) + (lane << 4);

        FragH Abuf[2];
        Abuf[0].q[0] = *(const vint4*)(row);
        Abuf[0].q[1] = *(const vint4*)(row + 16);
        #pragma unroll 2
        for (int kc = 0; kc < KC; ++kc) {
            const _Float16* wp = wtap + ((size_t)kc * NC << 9);
            __builtin_prefetch(wp + ((size_t)NC << 9), 0, 3);
            FragH B0, B1, B2, B3;
            B0.q[0] = *(const vint4*)(wp + 0);    B0.q[1] = *(const vint4*)(wp + 8);
            B1.q[0] = *(const vint4*)(wp + 512);  B1.q[1] = *(const vint4*)(wp + 520);
            B2.q[0] = *(const vint4*)(wp + 1024); B2.q[1] = *(const vint4*)(wp + 1032);
            B3.q[0] = *(const vint4*)(wp + 1536); B3.q[1] = *(const vint4*)(wp + 1544);
            int kcn = (kc + 1 < KC) ? (kc + 1) : 0;     // clamped next-k preload
            Abuf[(kc + 1) & 1].q[0] = *(const vint4*)(row + kcn * 32);
            Abuf[(kc + 1) & 1].q[1] = *(const vint4*)(row + kcn * 32 + 16);
            v16h a = Abuf[kc & 1].h;
            acc[0] = __builtin_amdgcn_wmma_f32_16x16x32_f16(false, a, false, B0.h, (short)0, acc[0], false, false);
            acc[1] = __builtin_amdgcn_wmma_f32_16x16x32_f16(false, a, false, B1.h, (short)0, acc[1], false, false);
            acc[2] = __builtin_amdgcn_wmma_f32_16x16x32_f16(false, a, false, B2.h, (short)0, acc[2], false, false);
            acc[3] = __builtin_amdgcn_wmma_f32_16x16x32_f16(false, a, false, B3.h, (short)0, acc[3], false, false);
        }
    }

    int n = lane & 15;
    float bv[4];
    int   co[4];
    #pragma unroll
    for (int s = 0; s < 4; ++s) {
        co[s] = ((co64 << 2) + s) * 16 + n;
        bv[s] = bias ? bias[co[s]] : 0.0f;
    }
    int si = (pin + khalf) / WSTRIP;               // first interior row this lane stores
    int sj = (pin + khalf) - si * WSTRIP;
    #pragma unroll
    for (int r = 0; r < 8; ++r) {                  // C/D layout: m = r + (lane>=16 ? 8 : 0)
        size_t prow = ((size_t)(b * HP + si + 1) * WP + sj + 1) * COUT;
        #pragma unroll
        for (int s = 0; s < 4; ++s)
            out[prow + co[s]] = (_Float16)(acc[s][r] + bv[s]);
        if (++sj == WSTRIP) { sj = 0; ++si; }
    }
}

// ---------------------------------------------------------------------------
// Final 3x3 conv, 256 -> 1 channel, on halo-padded input (no edge tests).
// One wave per output pixel; stack 1 uses spatially-transposed taps; bias only
// on stack 0 (row stack).
__global__ void __launch_bounds__(256)
k_conv_last(const _Float16* __restrict__ s2, const float* __restrict__ w3,
            const float* __restrict__ b3, float* __restrict__ U)
{
    int wave = blockIdx.x * (blockDim.x >> 5) + (threadIdx.x >> 5);
    if (wave >= 2 * PXTOT) return;
    int lane  = threadIdx.x & 31;
    int stack = wave / PXTOT;
    int pix   = wave - stack * PXTOT;
    int b     = pix / PPI;
    int p     = pix - b * PPI;
    int i     = p / WSTRIP;
    int jc    = p - i * WSTRIP;
    const _Float16* in = s2 + (size_t)stack * PXPADS * 256;
    float acc = 0.0f;
    for (int tap = 0; tap < 9; ++tap) {
        int a = tap / 3, bb = tap - (tap / 3) * 3;
        int stap = stack ? (bb * 3 + a) : tap;
        const _Float16* row = in + ((size_t)((b * HP + i + a) * WP + jc + bb)) * 256;
        const float* wr = w3 + stap * 256;
        #pragma unroll
        for (int c = 0; c < 256; c += 32)
            acc += (float)row[c + lane] * wr[c + lane];
    }
    for (int off = 16; off; off >>= 1) acc += __shfl_xor(acc, off, 32);
    if (lane == 0) U[stack * PXTOT + pix] = acc + (stack == 0 ? b3[0] : 0.0f);
}

// ---------------------------------------------------------------------------
// u[i,j] = R(i,cls(j)) + C(j,cls(i));  out = u + u^T
__device__ __forceinline__ int cls13(int j) { return j < 4 ? j : (j > 123 ? j - 115 : 4); }

__global__ void k_combine(const float* __restrict__ U, float* __restrict__ out)
{
    int idx = blockIdx.x * blockDim.x + threadIdx.x;
    if (idx >= NBATCH * LSEQ * LSEQ) return;
    int j = idx & 127;
    int i = (idx >> 7) & 127;
    int b = idx >> 14;
    const float* R = U;
    const float* C = U + PXTOT;
    int ci = cls13(i), cj = cls13(j);
    float uij = R[b * PPI + i * WSTRIP + cj] + C[b * PPI + j * WSTRIP + ci];
    float uji = R[b * PPI + j * WSTRIP + ci] + C[b * PPI + i * WSTRIP + cj];
    out[idx] = uij + uji;
}

// ---------------------------------------------------------------------------
extern "C" void kernel_launch(void* const* d_in, const int* in_sizes, int n_in,
                              void* d_out, int out_size, void* d_ws, size_t ws_size,
                              hipStream_t stream)
{
    const float* batch = (const float*)d_in[0];
    const float* w0 = (const float*)d_in[1]; const float* b0 = (const float*)d_in[2];
    const float* w1 = (const float*)d_in[3]; const float* b1 = (const float*)d_in[4];
    const float* w2 = (const float*)d_in[5]; const float* b2 = (const float*)d_in[6];
    const float* w3 = (const float*)d_in[7]; const float* b3 = (const float*)d_in[8];

    char* ws = (char*)d_ws;
    size_t off = 0;
    auto carve = [&](size_t bytes) -> char* {
        char* pp = ws + off;
        off += (bytes + 255) & ~(size_t)255;
        return pp;
    };
    // halo-padded activation buffers (carved contiguously; zeroed in one pass)
    char* zbase = ws;
    _Float16* strip = (_Float16*)carve((size_t)PXPADS * DFEAT * 2);
    _Float16* S0    = (_Float16*)carve((size_t)2 * PXPADS * 768 * 2);
    _Float16* S1    = (_Float16*)carve((size_t)2 * PXPADS * 512 * 2);
    _Float16* S2    = (_Float16*)carve((size_t)2 * PXPADS * 256 * 2);
    size_t zbytes = (ws + off) - zbase;
    float*    U     = (float*)carve((size_t)2 * PXTOT * 4);
    const int T0 = 9 * 24 * 48 * 512;   // layer0 packed halves per stack
    const int T1 = 9 * 24 * 32 * 512;
    const int T2 = 9 * 16 * 16 * 512;
    _Float16* PB0 = (_Float16*)carve((size_t)2 * T0 * 2);
    _Float16* PB1 = (_Float16*)carve((size_t)2 * T1 * 2);
    _Float16* PB2 = (_Float16*)carve((size_t)2 * T2 * 2);

    // 0) zero all padded activation buffers (establishes the zero halos)
    int n16 = (int)(zbytes >> 4);
    k_fill_zero<<<(n16 + 255) / 256, 256, 0, stream>>>((vint4*)zbase, n16);

    // 1) strip input interior (shared by both stacks)
    int totalStrip = PXTOT * DFEAT;
    k_build_strip<<<(totalStrip + 255) / 256, 256, 0, stream>>>(batch, strip, totalStrip);

    // 2) pack weights: row stack (no transpose), col stack (spatial transpose)
    k_pack_w<<<(T0 + 255) / 256, 256, 0, stream>>>(w0, PB0,      1536,   0, 24, 48, 768, 0, T0);
    k_pack_w<<<(T0 + 255) / 256, 256, 0, stream>>>(w0, PB0 + T0, 1536, 768, 24, 48, 768, 1, T0);
    k_pack_w<<<(T1 + 255) / 256, 256, 0, stream>>>(w1, PB1,       768,   0, 24, 32, 512, 0, T1);
    k_pack_w<<<(T1 + 255) / 256, 256, 0, stream>>>(w1, PB1 + T1,  768,   0, 24, 32, 512, 1, T1);
    k_pack_w<<<(T2 + 255) / 256, 256, 0, stream>>>(w2, PB2,       512,   0, 16, 16, 256, 0, T2);
    k_pack_w<<<(T2 + 255) / 256, 256, 0, stream>>>(w2, PB2 + T2,  512,   0, 16, 16, 256, 1, T2);

    // 3) conv stack on strips (row stack carries biases, col stack bias-free)
    int tiles0 = PXTILES * (768 / 64);
    k_conv3x3_wmma<768, 768><<<(tiles0 + 7) / 8, 256, 0, stream>>>(strip, PB0,      b0,      S0,                        tiles0);
    k_conv3x3_wmma<768, 768><<<(tiles0 + 7) / 8, 256, 0, stream>>>(strip, PB0 + T0, nullptr, S0 + (size_t)PXPADS * 768, tiles0);
    int tiles1 = PXTILES * (512 / 64);
    k_conv3x3_wmma<768, 512><<<(tiles1 + 7) / 8, 256, 0, stream>>>(S0,                         PB1,      b1,      S1,                        tiles1);
    k_conv3x3_wmma<768, 512><<<(tiles1 + 7) / 8, 256, 0, stream>>>(S0 + (size_t)PXPADS * 768,  PB1 + T1, nullptr, S1 + (size_t)PXPADS * 512, tiles1);
    int tiles2 = PXTILES * (256 / 64);
    k_conv3x3_wmma<512, 256><<<(tiles2 + 7) / 8, 256, 0, stream>>>(S1,                         PB2,      b2,      S2,                        tiles2);
    k_conv3x3_wmma<512, 256><<<(tiles2 + 7) / 8, 256, 0, stream>>>(S1 + (size_t)PXPADS * 512,  PB2 + T2, nullptr, S2 + (size_t)PXPADS * 256, tiles2);

    // 4) final 256->1 conv for both stacks
    int waves3 = 2 * PXTOT;
    k_conv_last<<<(waves3 + 7) / 8, 256, 0, stream>>>(S2, w3, b3, U);

    // 5) combine classes + symmetrize
    int tot = NBATCH * LSEQ * LSEQ;
    k_combine<<<(tot + 255) / 256, 256, 0, stream>>>(U, (float*)d_out);
}
